// multi_graph_14345190769255
// MI455X (gfx1250) — matrix operand order repaired and checked by reading, not compiled, not run on hardware
//
#include <hip/hip_runtime.h>

// ---------------------------------------------------------------------------
// Problem constants (match reference): N=100000, F=256, E=128, B=10000, D=32
// ---------------------------------------------------------------------------
#define F_DIM 256
#define E_DIM 128
#define D_NBR 32

typedef _Float16 v8h  __attribute__((ext_vector_type(8)));
typedef _Float16 v16h __attribute__((ext_vector_type(16)));
typedef float    v8f  __attribute__((ext_vector_type(8)));

static __device__ __forceinline__ v16h hcat(v8h lo, v8h hi) {
  return __builtin_shufflevector(lo, hi, 0,1,2,3,4,5,6,7,8,9,10,11,12,13,14,15);
}

static __device__ __forceinline__ v8f wmma16(v16h a, v16h b, v8f c) {
  // D = A(16x32 f16) * B(32x16 f16) + C(16x16 f32)
  return __builtin_amdgcn_wmma_f32_16x16x32_f16(false, a, false, b, (short)0, c,
                                                false, false);
}

static __device__ __forceinline__ v8f zero8() {
  v8f z = {0.f,0.f,0.f,0.f,0.f,0.f,0.f,0.f};
  return z;
}

// ---------------------------------------------------------------------------
// Kernel 1: build f16, transposed weights in workspace.
//   ws[0      .. 32767] : WeT  [e][f]  = W[f][e] * z(f)      (K=256 per row)
//   ws[32768  .. 65535] : A1T  [e][k]  = att1_w[k][e]        (K=256 per row)
//   ws[65536  .. 81919] : A2T  [e][k]  = att2_w[k][e]        (K=128 per row)
// ---------------------------------------------------------------------------
__global__ void prep_weights(const float* __restrict__ W,
                             const float* __restrict__ qz,
                             const float* __restrict__ a1w,
                             const float* __restrict__ a2w,
                             _Float16* __restrict__ ws) {
  int idx = blockIdx.x * blockDim.x + threadIdx.x;
  _Float16* wsWeT = ws;
  _Float16* wsA1  = ws + E_DIM * F_DIM;
  _Float16* wsA2  = ws + 2 * E_DIM * F_DIM;
  if (idx < E_DIM * F_DIM) {
    int e = idx >> 8;          // 0..127
    int f = idx & 255;         // 0..255
    float sig = 1.f / (1.f + __expf(-qz[f]));
    float z = sig * 1.2f - 0.1f;               // ZETA-GAMMA = 1.2, GAMMA = -0.1
    z = fminf(fmaxf(z, 0.f), 1.f);
    wsWeT[idx] = (_Float16)(W[f * E_DIM + e] * z);
    wsA1[idx]  = (_Float16)(a1w[f * E_DIM + e]);   // att1_w is [2E=256][E=128]
  }
  if (idx < E_DIM * E_DIM) {
    int e = idx >> 7;
    int k = idx & 127;
    wsA2[idx] = (_Float16)(a2w[k * E_DIM + e]);
  }
}

// ---------------------------------------------------------------------------
// Kernel 2: fully fused per-batch-element pipeline. One WG (8 waves) per b.
// ---------------------------------------------------------------------------
#define SXA 264   // f16 A-staging row stride (256 + 8 pad -> conflict-free b128)
#define SXB 136   // f16 A-staging row stride for K=128 (128 + 8 pad)
#define SCS 132   // f32 feat row stride (128 + 4 pad)
#define SHS 136   // f16 h2 row stride

__global__ __launch_bounds__(256)
void fused_gat(const int* __restrict__ nodes,
               const int* __restrict__ nidx,
               const int* __restrict__ nmask,
               const float* __restrict__ feat,
               const float* __restrict__ bE_g,
               const float* __restrict__ b1_g,
               const float* __restrict__ b2_g,
               const float* __restrict__ w3_g,
               const float* __restrict__ b3_g,
               const _Float16* __restrict__ ws,
               float* __restrict__ out) {
  __shared__ _Float16 sXa[48 * SXA];     // 25344 B : GEMM A staging (K=256)
  __shared__ _Float16 sXb[32 * SXB];     //  8704 B : GEMM A staging (K=128)
  __shared__ float    sC [33 * SCS];     // 17424 B : neigh_feat rows 0-31, node row 32
  __shared__ _Float16 sH [32 * SHS];     //  8704 B : h2 activations
  __shared__ float    sAtt3[E_DIM];
  __shared__ float    sPart[256];
  __shared__ float    sScore[D_NBR];
  __shared__ int      sFlag;

  const int b    = blockIdx.x;
  const int tid  = threadIdx.x;
  const int lane = tid & 31;
  const int wv   = tid >> 5;     // wave id 0..7 == N-tile
  const int lm   = lane & 15;
  const int lh   = lane >> 4;    // 0 or 1

  const _Float16* wsWeT = ws;
  const _Float16* wsA1  = ws + E_DIM * F_DIM;
  const _Float16* wsA2  = ws + 2 * E_DIM * F_DIM;

  // ---- Stage gathered rows (32 neighbors + node) as f16 into sXa ----------
  const int nodeRow = nodes[b];
  for (int c = tid; c < 33 * (F_DIM / 8); c += 256) {
    int r  = c >> 5;                 // row 0..32
    int k8 = (c & 31) * 8;           // K offset, multiple of 8
    int row = (r < 32) ? nidx[b * D_NBR + r] : nodeRow;
    const float4* s4 = (const float4*)(feat + (long)row * F_DIM + k8);
    float4 x0 = s4[0], x1 = s4[1];
    v8h h;
    h[0]=(_Float16)x0.x; h[1]=(_Float16)x0.y; h[2]=(_Float16)x0.z; h[3]=(_Float16)x0.w;
    h[4]=(_Float16)x1.x; h[5]=(_Float16)x1.y; h[6]=(_Float16)x1.z; h[7]=(_Float16)x1.w;
    *(v8h*)&sXa[r * SXA + k8] = h;
  }
  for (int c = tid; c < 15 * (F_DIM / 8); c += 256) {   // zero pad rows 33..47
    int r = 33 + (c >> 5);
    int k8 = (c & 31) * 8;
    v8h z = {};
    *(v8h*)&sXa[r * SXA + k8] = z;
  }
  if (tid < E_DIM) sAtt3[tid] = w3_g[tid];
  __syncthreads();

  // ---- GEMM1: [48,256] x WeT -> feats [48,128]; keep rows 0..32 -----------
  {
    float biasE = bE_g[wv * 16 + lm];
    v8f acc[3]; acc[0] = zero8(); acc[1] = zero8(); acc[2] = zero8();
#pragma unroll
    for (int ks = 0; ks < 8; ++ks) {
      const _Float16* bp = wsWeT + (wv * 16 + lm) * F_DIM + ks * 32 + lh * 8;
      v16h bf = hcat(*(const v8h*)bp, *(const v8h*)(bp + 16));
#pragma unroll
      for (int mt = 0; mt < 3; ++mt) {
        const _Float16* ap = &sXa[(mt * 16 + lm) * SXA + ks * 32 + lh * 8];
        v16h af = hcat(*(const v8h*)ap, *(const v8h*)(ap + 16));
        acc[mt] = wmma16(af, bf, acc[mt]);
      }
    }
#pragma unroll
    for (int mt = 0; mt < 3; ++mt)
#pragma unroll
      for (int r = 0; r < 8; ++r) {
        int row = mt * 16 + lh * 8 + r;       // C layout: M = r + 8*lh per tile
        if (row < 33)
          sC[row * SCS + wv * 16 + lm] = acc[mt][r] + biasE;
      }
  }
  __syncthreads();

  // ---- Stage X1 = concat(neigh_feat[d], node_feat) as f16 [32,256] --------
  for (int idx = tid; idx < 32 * F_DIM; idx += 256) {
    int d = idx >> 8;
    int k = idx & 255;
    float v = (k < E_DIM) ? sC[d * SCS + k] : sC[32 * SCS + (k - E_DIM)];
    sXa[d * SXA + k] = (_Float16)v;
  }
  __syncthreads();

  // ---- GEMM2: h1 = relu(X1 @ att1_w + b1), stage f16 into sXb -------------
  {
    float bias1 = b1_g[wv * 16 + lm];
    v8f acc[2]; acc[0] = zero8(); acc[1] = zero8();
#pragma unroll
    for (int ks = 0; ks < 8; ++ks) {
      const _Float16* bp = wsA1 + (wv * 16 + lm) * F_DIM + ks * 32 + lh * 8;
      v16h bf = hcat(*(const v8h*)bp, *(const v8h*)(bp + 16));
#pragma unroll
      for (int mt = 0; mt < 2; ++mt) {
        const _Float16* ap = &sXa[(mt * 16 + lm) * SXA + ks * 32 + lh * 8];
        v16h af = hcat(*(const v8h*)ap, *(const v8h*)(ap + 16));
        acc[mt] = wmma16(af, bf, acc[mt]);
      }
    }
#pragma unroll
    for (int mt = 0; mt < 2; ++mt)
#pragma unroll
      for (int r = 0; r < 8; ++r) {
        int row = mt * 16 + lh * 8 + r;
        float v = fmaxf(acc[mt][r] + bias1, 0.f);
        sXb[row * SXB + wv * 16 + lm] = (_Float16)v;
      }
  }
  __syncthreads();

  // ---- GEMM3: h2 = relu(h1 @ att2_w + b2) ---------------------------------
  {
    float bias2 = b2_g[wv * 16 + lm];
    v8f acc[2]; acc[0] = zero8(); acc[1] = zero8();
#pragma unroll
    for (int ks = 0; ks < 4; ++ks) {
      const _Float16* bp = wsA2 + (wv * 16 + lm) * E_DIM + ks * 32 + lh * 8;
      v16h bf = hcat(*(const v8h*)bp, *(const v8h*)(bp + 16));
#pragma unroll
      for (int mt = 0; mt < 2; ++mt) {
        const _Float16* ap = &sXb[(mt * 16 + lm) * SXB + ks * 32 + lh * 8];
        v16h af = hcat(*(const v8h*)ap, *(const v8h*)(ap + 16));
        acc[mt] = wmma16(af, bf, acc[mt]);
      }
    }
#pragma unroll
    for (int mt = 0; mt < 2; ++mt)
#pragma unroll
      for (int r = 0; r < 8; ++r) {
        int row = mt * 16 + lh * 8 + r;
        float v = fmaxf(acc[mt][r] + bias2, 0.f);
        sH[row * SHS + wv * 16 + lm] = (_Float16)v;
      }
  }
  __syncthreads();

  // ---- scores[d] = h2[d] . att3_w + b3 (deterministic tree reduction) -----
  {
    int d = tid >> 3;         // pair 0..31
    int seg = tid & 7;        // 8 threads per pair, 16 cols each
    float p = 0.f;
#pragma unroll
    for (int j = 0; j < 16; ++j) {
      int n = seg * 16 + j;
      p += (float)sH[d * SHS + n] * sAtt3[n];
    }
    sPart[tid] = p;
  }
  __syncthreads();

  // ---- masked softmax over D=32 within wave 0 -----------------------------
  if (wv == 0) {
    float s = b3_g[0];
#pragma unroll
    for (int g = 0; g < 8; ++g) s += sPart[lane * 8 + g];
    int mk = nmask[b * D_NBR + lane];
    int has = __any(mk != 0);
    s = mk ? s : -1e30f;
    float m = s;
#pragma unroll
    for (int off = 16; off >= 1; off >>= 1)
      m = fmaxf(m, __shfl_xor(m, off, 32));
    float e = __expf(s - m);
    float sum = e;
#pragma unroll
    for (int off = 16; off >= 1; off >>= 1)
      sum += __shfl_xor(sum, off, 32);
    sScore[lane] = e / sum;
    if (lane == 0) sFlag = has ? 1 : 0;
  }
  __syncthreads();

  // ---- agg = sum_d att[d] * neigh_feat[d]; fallback to node_feat ----------
  if (tid < E_DIM) {
    float agg = 0.f;
#pragma unroll
    for (int d = 0; d < D_NBR; ++d)
      agg += sScore[d] * sC[d * SCS + tid];
    float res = sFlag ? agg : sC[32 * SCS + tid];
    out[(long)b * E_DIM + tid] = res;
  }
}

// ---------------------------------------------------------------------------
extern "C" void kernel_launch(void* const* d_in, const int* in_sizes, int n_in,
                              void* d_out, int out_size, void* d_ws, size_t ws_size,
                              hipStream_t stream) {
  const int*   nodes = (const int*)d_in[0];
  const int*   nidx  = (const int*)d_in[1];
  const int*   nmask = (const int*)d_in[2];
  const float* feat  = (const float*)d_in[3];
  const float* W     = (const float*)d_in[4];
  const float* bE    = (const float*)d_in[5];
  const float* qz    = (const float*)d_in[6];
  const float* a1w   = (const float*)d_in[7];
  const float* a1b   = (const float*)d_in[8];
  const float* a2w   = (const float*)d_in[9];
  const float* a2b   = (const float*)d_in[10];
  const float* a3w   = (const float*)d_in[11];
  const float* a3b   = (const float*)d_in[12];
  float*       outp  = (float*)d_out;
  _Float16*    ws    = (_Float16*)d_ws;

  const int B = in_sizes[0];   // 10000

  prep_weights<<<(E_DIM * F_DIM + 255) / 256, 256, 0, stream>>>(W, qz, a1w, a2w, ws);
  fused_gat<<<B, 256, 0, stream>>>(nodes, nidx, nmask, feat, bE, a1b, a2b,
                                   a3w, a3b, ws, outp);
}
